// EmbedLayerUpdate_24678882083059
// MI455X (gfx1250) — matrix operand order repaired and checked by reading, not compiled, not run on hardware
//
#include <hip/hip_runtime.h>

typedef __attribute__((ext_vector_type(16))) __bf16 v16bf;
typedef __attribute__((ext_vector_type(8)))  __bf16 v8bf;
typedef __attribute__((ext_vector_type(8)))  float  v8f;
typedef __attribute__((ext_vector_type(4)))  int    v4i;

#define P_DIM 64
#define B_DIM 16
#define N_DIM 4096
#define NODES (B_DIM * N_DIM)          /* 65536 */
#define MU_ELEMS (P_DIM * NODES)       /* 4194304 */

// ---- CDNA5 async-to-LDS path (probe via __has_builtin; safe fallback) ------
#if defined(__has_builtin)
# if __has_builtin(__builtin_amdgcn_global_load_async_to_lds_b128)
#  define USE_ASYNC_LDS 1
# endif
#endif
#ifndef USE_ASYNC_LDS
# define USE_ASYNC_LDS 0
#endif

// builtin signature: (AS1 v4i* gsrc, AS3 v4i* ldst, int offset, int cpol)
#define AS1_V4I(p) ((__attribute__((address_space(1))) v4i*)(const void*)(p))
#define AS3_V4I(p) ((__attribute__((address_space(3))) v4i*)(p))

static __device__ __forceinline__ void wait_async0() {
#if defined(__has_builtin) && __has_builtin(__builtin_amdgcn_s_wait_asynccnt)
  __builtin_amdgcn_s_wait_asynccnt(0);
#else
  asm volatile("s_wait_asynccnt 0" ::: "memory");
#endif
}

static __device__ __forceinline__ unsigned short f2bf(float f) {
  unsigned int u = __float_as_uint(f);
  u += 0x7FFFu + ((u >> 16) & 1u);     // round-to-nearest-even
  return (unsigned short)(u >> 16);
}
static __device__ __forceinline__ unsigned int pk2(float a, float b) {
  return (unsigned int)f2bf(a) | ((unsigned int)f2bf(b) << 16);
}
// A-fragment (16x32 bf16): lane holds M=lane&15; K = (lane>>4)*16 .. +15 (contiguous)
static __device__ __forceinline__ v16bf load_a_frag(const unsigned short* p) {
  v8bf lo = *reinterpret_cast<const v8bf*>(p);
  v8bf hi = *reinterpret_cast<const v8bf*>(p + 8);
  return __builtin_shufflevector(lo, hi, 0,1,2,3,4,5,6,7,8,9,10,11,12,13,14,15);
}

// ---------------------------------------------------------------------------
// Per-variable node embedding MLP: relu(W2 @ relu(W1 @ f + b1) + b2)
// Layer-2 (64x64 @ 64xnodes) mapped onto v_wmma_f32_16x16x32_bf16.
// ---------------------------------------------------------------------------
__global__ __launch_bounds__(256)
void embed_mlp_kernel(const float* __restrict__ image, const float* __restrict__ lbs,
                      const float* __restrict__ ubs, const float* __restrict__ rhos,
                      const float* __restrict__ tiW0, const float* __restrict__ tiW1,
                      const float* __restrict__ tib0, const float* __restrict__ tib1,
                      const float* __restrict__ tW0, const float* __restrict__ tW1,
                      const float* __restrict__ tb0, const float* __restrict__ tb1,
                      float* __restrict__ mu_out, unsigned short* __restrict__ mu_bf)
{
  __shared__ __align__(64) float F[3][128];
  __shared__ __align__(64) unsigned short Ht[128][64];  // (node, k) bf16
  __shared__ __align__(64) unsigned short At[64 * 64];  // W2 (p, k) bf16
  const int t = threadIdx.x;
  const int v = blockIdx.x >> 9;               // 512 node-tiles per variable
  const int tile0 = (blockIdx.x & 511) * 128;
  const float* W1 = (v == 0) ? tiW0 : tW0;
  const float* B1 = (v == 0) ? tib0 : tb0;
  const float* W2 = (v == 0) ? tiW1 : tW1;
  const float* B2 = (v == 0) ? tib1 : tb1;

  if (t < 128) {
    int node = tile0 + t;
    float f0, f1, f2;
    if (v == 0) { f0 = image[node]; f1 = lbs[node]; f2 = ubs[node]; }
    else {
      f0 = lbs[v * NODES + node];
      f1 = ubs[v * NODES + node];
      f2 = (v < 3) ? rhos[(v - 1) * NODES + node] : 0.0f;
    }
    F[0][t] = f0; F[1][t] = f1; F[2][t] = f2;
  }
  for (int i = t; i < 4096; i += 256) At[i] = f2bf(W2[i]);
  __syncthreads();

  // layer 1: 3 -> 64 on VALU, bf16 result to LDS (node-major for B operand)
  for (int i = t; i < 8192; i += 256) {
    int k = i >> 7, node = i & 127;
    float h = W1[k*3+0]*F[0][node] + W1[k*3+1]*F[1][node] + W1[k*3+2]*F[2][node] + B1[k];
    Ht[node][k] = f2bf(fmaxf(h, 0.0f));
  }
  __syncthreads();

  // layer 2: 64x64 @ 64x128 via WMMA; wave w owns 16 node-columns
  const int lane = t & 31, w = t >> 5;
  const int nloc = w * 16 + (lane & 15);
  v8f zero = {0.f,0.f,0.f,0.f,0.f,0.f,0.f,0.f};
  v8f acc[4];
  #pragma unroll
  for (int i = 0; i < 4; i++) acc[i] = zero;

  #pragma unroll
  for (int ks = 0; ks < 2; ks++) {
    v16bf bfrag = *reinterpret_cast<const v16bf*>(&Ht[nloc][ks*32 + (lane >> 4) * 16]);
    #pragma unroll
    for (int pf = 0; pf < 4; pf++) {
      v16bf afrag = load_a_frag(&At[(pf*16 + (lane & 15)) * 64 + ks*32 + (lane >> 4) * 16]);
      acc[pf] = __builtin_amdgcn_wmma_f32_16x16x32_bf16(
          false, afrag, false, bfrag, (short)0, acc[pf], false, false);
    }
  }
  const int node = tile0 + nloc;
  float vals[4][8];
  #pragma unroll
  for (int pf = 0; pf < 4; pf++) {
    #pragma unroll
    for (int r = 0; r < 8; r++) {
      int p = pf*16 + r + 8*(lane >> 4);
      float val = fmaxf(acc[pf][r] + B2[p], 0.0f);
      vals[pf][r] = val;
      mu_out[(size_t)v * MU_ELEMS + (size_t)p * NODES + node] = val;
    }
  }
  if (v == 0) {  // single branch: bf16 shadow (A operand for fwd layer 0)
    #pragma unroll
    for (int pf = 0; pf < 4; pf++) {
      #pragma unroll
      for (int r = 0; r < 8; r++) {
        int p = pf*16 + r + 8*(lane >> 4);
        mu_bf[(size_t)p * NODES + node] = f2bf(vals[pf][r]);
      }
    }
  }
}

// ---------------------------------------------------------------------------
// sums[m] = sum_n mu[m, n], m = p*16+b  (wave-per-row shuffle reduction)
// ---------------------------------------------------------------------------
__global__ __launch_bounds__(256)
void rowsum_kernel(const float* __restrict__ mu_v, float* __restrict__ sums) {
  int lane = threadIdx.x & 31, w = threadIdx.x >> 5;
  int row = blockIdx.x * 8 + w;
  const float* src = mu_v + (size_t)row * N_DIM;
  float s = 0.0f;
  for (int i = lane; i < N_DIM; i += 32) s += src[i];
  #pragma unroll
  for (int off = 16; off >= 1; off >>= 1) s += __shfl_down(s, off, 32);
  if (lane == 0) sums[row] = s;
}

// embed[q,b] = (pre * sum_p tW[q,p]*sums[p,b] + tb[q]) * post   (grid 4 x 256)
__global__ __launch_bounds__(256)
void theta_small_kernel(const float* __restrict__ sums, const float* __restrict__ tW,
                        const float* __restrict__ tb, float pre, float post,
                        float* __restrict__ embed) {
  int idx = blockIdx.x * 256 + threadIdx.x;   // q*16+b
  int q = idx >> 4, b = idx & 15;
  float s = 0.0f;
  for (int p = 0; p < 64; p++) s += tW[q * 64 + p] * sums[p * 16 + b];
  embed[idx] = (s * pre + tb[q]) * post;
}

// cvec[i] = sum_o b[o] * W[o,i]   (folds "mc - b" into the backward GEMM bias)
__global__ __launch_bounds__(256)
void cvec_kernel(const float* __restrict__ W, const float* __restrict__ bvec,
                 float* __restrict__ cvec) {
  int i = blockIdx.x * 256 + threadIdx.x;     // grid 16
  float s = 0.0f;
  for (int o = 0; o < N_DIM; o++) s += bvec[o] * W[(size_t)o * N_DIM + i];
  cvec[i] = s;
}

// ---------------------------------------------------------------------------
// Main GEMM: out[m, n] = sum_k Abf[m, k] * B[k, n] + bsign*bias[n]
//   wIsNK=1 (forward):  B[k][n] = W[n*4096 + k]   (fwd_lin, W transposed)
//   wIsNK=0 (backward): B[k][n] = W[k*4096 + n]   (bwd_lin)
// 128x128 block tile, BK=32, double-buffered LDS, 1 barrier per K-step.
// A (bf16) is moved with GLOBAL_LOAD_ASYNC_TO_LDS_B128 (ASYNCcnt); W (fp32)
// is staged through VGPRs with on-the-fly f32->bf16 conversion.
// ---------------------------------------------------------------------------
__global__ __launch_bounds__(256)
void gemm_bf16_kernel(const unsigned short* __restrict__ Abf, const float* __restrict__ W,
                      const float* __restrict__ bias, float bsign, int wIsNK,
                      float* __restrict__ out) {
  __shared__ __align__(64) unsigned short As[2][128][32];  // (m, k)
  __shared__ __align__(64) unsigned short Bs[2][128][32];  // (n, k) transposed stage
  const int t = threadIdx.x;
  const int lane = t & 31, w = t >> 5;
  const int wm = w >> 2, wn = w & 3;
  const int m0 = blockIdx.y * 128;
  const int n0 = blockIdx.x * 128;
  const int arow = t >> 1, akc = (t & 1) * 16;   // A/B(fwd) staging: 16 elems each
  const int bk = t >> 3,  bnc = (t & 7) * 16;    // B(bwd) staging

  const unsigned short* aga = Abf + (size_t)(m0 + arow) * N_DIM + akc;

  v8f zero = {0.f,0.f,0.f,0.f,0.f,0.f,0.f,0.f};
  v8f acc[4][2];
  #pragma unroll
  for (int i = 0; i < 4; i++) { acc[i][0] = zero; acc[i][1] = zero; }

#if !USE_ASYNC_LDS
  uint4 aReg0, aReg1;
#endif
  float fv[16];

  // ---- prologue: tile 0 ----
  {
#if USE_ASYNC_LDS
    __builtin_amdgcn_global_load_async_to_lds_b128(
        AS1_V4I(aga + 0), AS3_V4I(&As[0][arow][akc]), 0, 0);
    __builtin_amdgcn_global_load_async_to_lds_b128(
        AS1_V4I(aga + 8), AS3_V4I(&As[0][arow][akc + 8]), 0, 0);
#else
    const uint4* g = reinterpret_cast<const uint4*>(aga);
    aReg0 = g[0]; aReg1 = g[1];
#endif
    const float* wsrc = wIsNK ? (W + (size_t)(n0 + arow) * N_DIM + akc)
                              : (W + (size_t)bk * N_DIM + n0 + bnc);
    const float4* wg = reinterpret_cast<const float4*>(wsrc);
    #pragma unroll
    for (int q = 0; q < 4; q++) {
      float4 f = wg[q];
      fv[q*4+0] = f.x; fv[q*4+1] = f.y; fv[q*4+2] = f.z; fv[q*4+3] = f.w;
    }
#if !USE_ASYNC_LDS
    *reinterpret_cast<uint4*>(&As[0][arow][akc])     = aReg0;
    *reinterpret_cast<uint4*>(&As[0][arow][akc + 8]) = aReg1;
#endif
    if (wIsNK) {
      unsigned int pk[8];
      #pragma unroll
      for (int q = 0; q < 8; q++) pk[q] = pk2(fv[2*q], fv[2*q+1]);
      *reinterpret_cast<uint4*>(&Bs[0][arow][akc])     = make_uint4(pk[0],pk[1],pk[2],pk[3]);
      *reinterpret_cast<uint4*>(&Bs[0][arow][akc + 8]) = make_uint4(pk[4],pk[5],pk[6],pk[7]);
    } else {
      #pragma unroll
      for (int q = 0; q < 16; q++) Bs[0][bnc + q][bk] = f2bf(fv[q]);
    }
#if USE_ASYNC_LDS
    wait_async0();
#endif
    __syncthreads();
  }

  for (int i = 0; i < 128; i++) {
    const int cur = i & 1, nxt = cur ^ 1;
    const int k1 = (i + 1) * 32;
    const bool more = (i + 1) < 128;

    // issue prefetch of tile i+1 (safe: barrier above proved everyone is done
    // reading buffer `nxt`, which was consumed in iteration i-1)
    if (more) {
#if USE_ASYNC_LDS
      __builtin_amdgcn_global_load_async_to_lds_b128(
          AS1_V4I(aga + k1), AS3_V4I(&As[nxt][arow][akc]), 0, 0);
      __builtin_amdgcn_global_load_async_to_lds_b128(
          AS1_V4I(aga + k1 + 8), AS3_V4I(&As[nxt][arow][akc + 8]), 0, 0);
#else
      const uint4* g = reinterpret_cast<const uint4*>(aga + k1);
      aReg0 = g[0]; aReg1 = g[1];
#endif
      const float* wsrc = wIsNK ? (W + (size_t)(n0 + arow) * N_DIM + k1 + akc)
                                : (W + (size_t)(k1 + bk) * N_DIM + n0 + bnc);
      const float4* wg = reinterpret_cast<const float4*>(wsrc);
      #pragma unroll
      for (int q = 0; q < 4; q++) {
        float4 f = wg[q];
        fv[q*4+0] = f.x; fv[q*4+1] = f.y; fv[q*4+2] = f.z; fv[q*4+3] = f.w;
      }
      __builtin_prefetch(aga + k1 + 32, 0, 1);
      __builtin_prefetch(wsrc + (wIsNK ? 32 : 32 * N_DIM), 0, 1);
    }

    // ---- compute on buffer `cur`: 4 A-frags hoisted, 2 B-frags, 8 WMMAs ----
    {
      v16bf b0 = *reinterpret_cast<const v16bf*>(
          &Bs[cur][wn*32 + (lane & 15)][(lane >> 4) * 16]);
      v16bf b1 = *reinterpret_cast<const v16bf*>(
          &Bs[cur][wn*32 + 16 + (lane & 15)][(lane >> 4) * 16]);
      v16bf af[4];
      #pragma unroll
      for (int mf = 0; mf < 4; mf++)
        af[mf] = load_a_frag(&As[cur][wm*64 + mf*16 + (lane & 15)][(lane >> 4) * 16]);
      #pragma unroll
      for (int mf = 0; mf < 4; mf++) {
        acc[mf][0] = __builtin_amdgcn_wmma_f32_16x16x32_bf16(
            false, af[mf], false, b0, (short)0, acc[mf][0], false, false);
        acc[mf][1] = __builtin_amdgcn_wmma_f32_16x16x32_bf16(
            false, af[mf], false, b1, (short)0, acc[mf][1], false, false);
      }
    }

    // ---- publish tile i+1 into buffer `nxt` ----
    if (more) {
#if !USE_ASYNC_LDS
      *reinterpret_cast<uint4*>(&As[nxt][arow][akc])     = aReg0;
      *reinterpret_cast<uint4*>(&As[nxt][arow][akc + 8]) = aReg1;
#endif
      if (wIsNK) {
        unsigned int pk[8];
        #pragma unroll
        for (int q = 0; q < 8; q++) pk[q] = pk2(fv[2*q], fv[2*q+1]);
        *reinterpret_cast<uint4*>(&Bs[nxt][arow][akc])     = make_uint4(pk[0],pk[1],pk[2],pk[3]);
        *reinterpret_cast<uint4*>(&Bs[nxt][arow][akc + 8]) = make_uint4(pk[4],pk[5],pk[6],pk[7]);
      } else {
        #pragma unroll
        for (int q = 0; q < 16; q++) Bs[nxt][bnc + q][bk] = f2bf(fv[q]);
      }
#if USE_ASYNC_LDS
      wait_async0();
#endif
      __syncthreads();
    }
  }

  #pragma unroll
  for (int mf = 0; mf < 4; mf++) {
    #pragma unroll
    for (int nf = 0; nf < 2; nf++) {
      int col   = n0 + wn*32 + nf*16 + (lane & 15);
      int mbase = m0 + wm*64 + mf*16 + 8*(lane >> 4);
      float bb = bsign * bias[col];
      #pragma unroll
      for (int r = 0; r < 8; r++)
        out[(size_t)(mbase + r) * N_DIM + col] = acc[mf][nf][r] + bb;
    }
  }
}

// ---------------------------------------------------------------------------
// Fused theta_p (64x64 over P) + ReLU update of mu (and bf16 shadow copy):
//   ep[q,col] = sum_p tW[q,p]*lin[p,col] + tb[q]
//   mu[q,col] = relu(t1*mu[q,col] + t2*ep + t3*embed[q, b(col)])
// ---------------------------------------------------------------------------
__global__ __launch_bounds__(256)
void update_kernel(const float* __restrict__ lin, const float* __restrict__ tW,
                   const float* __restrict__ tb, const float* __restrict__ embed,
                   const float* __restrict__ tvec,
                   float* __restrict__ mu_tgt, unsigned short* __restrict__ mu_bf) {
  __shared__ __align__(64) unsigned short At[64 * 64];  // tW (q, p) bf16
  __shared__ __align__(64) unsigned short Bt[128][64];  // (col, p) bf16
  const int t = threadIdx.x;
  const int col0 = blockIdx.x * 128;
  for (int i = t; i < 4096; i += 256) At[i] = f2bf(tW[i]);
  for (int i = t; i < 8192; i += 256) {
    int p = i >> 7, c = i & 127;
    Bt[c][p] = f2bf(lin[(size_t)p * NODES + col0 + c]);
  }
  __syncthreads();

  const int lane = t & 31, w = t >> 5;
  const int nloc = w * 16 + (lane & 15);
  const float t1 = tvec[0], t2 = tvec[1], t3 = tvec[2];
  v8f zero = {0.f,0.f,0.f,0.f,0.f,0.f,0.f,0.f};
  v8f acc[4];
  #pragma unroll
  for (int i = 0; i < 4; i++) acc[i] = zero;

  #pragma unroll
  for (int ks = 0; ks < 2; ks++) {
    v16bf bfrag = *reinterpret_cast<const v16bf*>(&Bt[nloc][ks*32 + (lane >> 4) * 16]);
    #pragma unroll
    for (int qf = 0; qf < 4; qf++) {
      v16bf afrag = load_a_frag(&At[(qf*16 + (lane & 15)) * 64 + ks*32 + (lane >> 4) * 16]);
      acc[qf] = __builtin_amdgcn_wmma_f32_16x16x32_bf16(
          false, afrag, false, bfrag, (short)0, acc[qf], false, false);
    }
  }
  const int col = col0 + nloc;
  const int b = col >> 12;   // col = b*4096 + o
  #pragma unroll
  for (int qf = 0; qf < 4; qf++) {
    #pragma unroll
    for (int r = 0; r < 8; r++) {
      int q = qf*16 + r + 8*(lane >> 4);
      float ep  = acc[qf][r] + tb[q];
      float old = mu_tgt[(size_t)q * NODES + col];
      float val = fmaxf(t1 * old + t2 * ep + t3 * embed[q * 16 + b], 0.0f);
      mu_tgt[(size_t)q * NODES + col] = val;
      mu_bf[(size_t)q * NODES + col] = f2bf(val);
    }
  }
}

// ---------------------------------------------------------------------------
extern "C" void kernel_launch(void* const* d_in, const int* in_sizes, int n_in,
                              void* d_out, int out_size, void* d_ws, size_t ws_size,
                              hipStream_t stream) {
  const float* image = (const float*)d_in[0];
  const float* lbs   = (const float*)d_in[1];
  const float* ubs   = (const float*)d_in[2];
  const float* rhos  = (const float*)d_in[3];
  const float* tiW0  = (const float*)d_in[4];
  const float* tiW1  = (const float*)d_in[5];
  const float* tib0  = (const float*)d_in[6];
  const float* tib1  = (const float*)d_in[7];
  const float* tW0   = (const float*)d_in[8];
  const float* tW1   = (const float*)d_in[9];
  const float* tb0   = (const float*)d_in[10];
  const float* tb1   = (const float*)d_in[11];
  const float* tempW[4] = {(const float*)d_in[12], (const float*)d_in[13],
                           (const float*)d_in[14], (const float*)d_in[15]};
  const float* tempB[4] = {(const float*)d_in[16], (const float*)d_in[17],
                           (const float*)d_in[18], (const float*)d_in[19]};
  const float* testF = (const float*)d_in[20];
  const float* testB = (const float*)d_in[21];
  const float* netW  = (const float*)d_in[22];
  const float* netb  = (const float*)d_in[23];
  float* out = (float*)d_out;

  char* ws = (char*)d_ws;
  unsigned short* mu_bf = (unsigned short*)ws;          // 8 MiB: bf16 shadow of current mu
  float* lin    = (float*)(ws + 8388608);               // 16 MiB: fwd_lin / bwd_lin
  float* sums   = (float*)(ws + 25165824);              // 4 KiB: (p,b) row sums
  float* embedv = (float*)(ws + 25169920);              // 4 KiB: embed (q,b)
  float* cvec   = (float*)(ws + 25174016);              // 16 KiB: b @ W for backward

  // mu[v] node embeddings (writes d_out in place; bf16 shadow of mu[0])
  embed_mlp_kernel<<<2048, 256, 0, stream>>>(image, lbs, ubs, rhos,
      tiW0, tiW1, tib0, tib1, tW0, tW1, tb0, tb1, out, mu_bf);

  dim3 gg(32, 8);  // (N/128, M/128)
  for (int j = 0; j < 3; j++) {
    rowsum_kernel<<<128, 256, 0, stream>>>(out + (size_t)j * MU_ELEMS, sums);
    theta_small_kernel<<<4, 256, 0, stream>>>(sums, tempW[0], tempB[0],
                                              1.0f / 4096.0f, 1.0f, embedv);
    gemm_bf16_kernel<<<gg, 256, 0, stream>>>(mu_bf, netW + (size_t)j * 16777216,
                                             netb + j * 4096, 1.0f, 1, lin);
    update_kernel<<<512, 256, 0, stream>>>(lin, tempW[1], tempB[1], embedv, testF,
                                           out + (size_t)(j + 1) * MU_ELEMS, mu_bf);
  }
  for (int j = 2; j >= 0; j--) {
    rowsum_kernel<<<128, 256, 0, stream>>>(out + (size_t)(j + 1) * MU_ELEMS, sums);
    theta_small_kernel<<<4, 256, 0, stream>>>(sums, tempW[2], tempB[2],
                                              1.0f, 1.0f / 4096.0f, embedv);
    cvec_kernel<<<16, 256, 0, stream>>>(netW + (size_t)j * 16777216, netb + j * 4096, cvec);
    gemm_bf16_kernel<<<gg, 256, 0, stream>>>(mu_bf, netW + (size_t)j * 16777216,
                                             cvec, -1.0f, 0, lin);
    update_kernel<<<512, 256, 0, stream>>>(lin, tempW[3], tempB[3], embedv, testB,
                                           out + (size_t)j * MU_ELEMS, mu_bf);
  }
}
